// QuantBertSelfOutput_37726992728250
// MI455X (gfx1250) — 8-bit matrix operand order repaired and checked by reading, compiled, not run on hardware
//
#include <hip/hip_runtime.h>
#include <hip/hip_bf16.h>

typedef int v8i __attribute__((ext_vector_type(8)));

#define QMAX 127.0f
#define LN_EPS 1e-12f

__global__ void init_ws_kernel(unsigned* absbits) {
    if (threadIdx.x == 0 && blockIdx.x == 0) absbits[0] = 0u;
}

__global__ void absmax_kernel(const float* __restrict__ x, size_t n,
                              unsigned* __restrict__ absbits) {
    size_t stride = (size_t)gridDim.x * blockDim.x * 4;
    float m = 0.0f;
    for (size_t i = ((size_t)blockIdx.x * blockDim.x + threadIdx.x) * 4;
         i + 3 < n; i += stride) {
        float4 v = *(const float4*)(x + i);
        m = fmaxf(m, fmaxf(fmaxf(fabsf(v.x), fabsf(v.y)),
                           fmaxf(fabsf(v.z), fabsf(v.w))));
    }
    #pragma unroll
    for (int o = 16; o > 0; o >>= 1)
        m = fmaxf(m, __shfl_xor(m, o, 32));
    if ((threadIdx.x & 31) == 0)
        atomicMax(absbits, __float_as_uint(m));
}

__global__ void quantw_kernel(const float* __restrict__ W, int H,
                              signed char* __restrict__ Wq,
                              float* __restrict__ wscale) {
    const int wave = threadIdx.x >> 5;
    const int lane = threadIdx.x & 31;
    const int o = blockIdx.x * 8 + wave;
    const float* wr = W + (size_t)o * H;

    float m = 0.0f;
    for (int c = lane * 4; c < H; c += 128) {
        float4 v = *(const float4*)(wr + c);
        m = fmaxf(m, fmaxf(fmaxf(fabsf(v.x), fabsf(v.y)),
                           fmaxf(fabsf(v.z), fabsf(v.w))));
    }
    #pragma unroll
    for (int oo = 16; oo > 0; oo >>= 1)
        m = fmaxf(m, __shfl_xor(m, oo, 32));

    const float sc = m / QMAX + 1e-8f;
    if (lane == 0) wscale[o] = sc;
    const float inv = 1.0f / sc;

    for (int c = lane * 4; c < H; c += 128) {
        float4 v = *(const float4*)(wr + c);
        int q0 = (int)fminf(fmaxf(rintf(v.x * inv), -128.0f), 127.0f);
        int q1 = (int)fminf(fmaxf(rintf(v.y * inv), -128.0f), 127.0f);
        int q2 = (int)fminf(fmaxf(rintf(v.z * inv), -128.0f), 127.0f);
        int q3 = (int)fminf(fmaxf(rintf(v.w * inv), -128.0f), 127.0f);
        int packed = (q0 & 255) | ((q1 & 255) << 8) | ((q2 & 255) << 16)
                   | (q3 << 24);
        *(int*)(Wq + (size_t)o * H + c) = packed;
    }
}

__global__ void __launch_bounds__(256)
gemm_ln_kernel(const float* __restrict__ x, const float* __restrict__ res,
               const signed char* __restrict__ Wq,
               const float* __restrict__ wscale,
               const float* __restrict__ bias,
               const float* __restrict__ gamma, const float* __restrict__ beta,
               const unsigned* __restrict__ absbits,
               float* __restrict__ out, int M, int H) {
    extern __shared__ char smem_raw[];
    signed char* smemA = (signed char*)smem_raw;
    float* zsm = (float*)smem_raw;
    const int LNS = H + 4;

    const int tid  = threadIdx.x;
    const int wave = tid >> 5;
    const int lane = tid & 31;
    const int m0   = blockIdx.x * 32;
    const int n0   = wave * 128;

    const float a_scale = __uint_as_float(absbits[0]) / QMAX + 1e-8f;
    const float inv_a   = 1.0f / a_scale;

    for (int i = tid * 4; i < 32 * H; i += 256 * 4) {
        float4 v = *(const float4*)(x + (size_t)m0 * H + i);
        int q0 = (int)fminf(fmaxf(rintf(v.x * inv_a), -128.0f), 127.0f);
        int q1 = (int)fminf(fmaxf(rintf(v.y * inv_a), -128.0f), 127.0f);
        int q2 = (int)fminf(fmaxf(rintf(v.z * inv_a), -128.0f), 127.0f);
        int q3 = (int)fminf(fmaxf(rintf(v.w * inv_a), -128.0f), 127.0f);
        *(int*)(smemA + i) = (q0 & 255) | ((q1 & 255) << 8)
                           | ((q2 & 255) << 16) | (q3 << 24);
    }
    __syncthreads();

    v8i acc0[8], acc1[8];
    #pragma unroll
    for (int t = 0; t < 8; ++t)
        #pragma unroll
        for (int r = 0; r < 8; ++r) { acc0[t][r] = 0; acc1[t][r] = 0; }

    const int lmod = lane & 15;
    const int lhi  = lane >> 4;

    for (int k0 = 0; k0 < H; k0 += 64) {
        const int abase = k0 + (lhi << 4);
        const int aoff0 = lmod * H + abase;
        const int aoff1 = (lmod + 16) * H + abase;
        int4 p0, p1;
        v8i a0, a1;
        p0 = *(const int4*)(smemA + aoff0 +  0);
        p1 = *(const int4*)(smemA + aoff0 + 32);
        a0[0] = p0.x; a0[1] = p0.y; a0[2] = p0.z; a0[3] = p0.w;
        a0[4] = p1.x; a0[5] = p1.y; a0[6] = p1.z; a0[7] = p1.w;
        p0 = *(const int4*)(smemA + aoff1 +  0);
        p1 = *(const int4*)(smemA + aoff1 + 32);
        a1[0] = p0.x; a1[1] = p0.y; a1[2] = p0.z; a1[3] = p0.w;
        a1[4] = p1.x; a1[5] = p1.y; a1[6] = p1.z; a1[7] = p1.w;

        #pragma unroll
        for (int t = 0; t < 8; ++t) {
            const signed char* bp =
                Wq + (size_t)(n0 + t * 16 + lmod) * H + k0 + (lhi << 4);
            int4 b0 = *(const int4*)bp;
            int4 b1 = *(const int4*)(bp + 32);
            v8i b;
            b[0] = b0.x; b[1] = b0.y; b[2] = b0.z; b[3] = b0.w;
            b[4] = b1.x; b[5] = b1.y; b[6] = b1.z; b[7] = b1.w;
            acc0[t] = __builtin_amdgcn_wmma_i32_16x16x64_iu8(
                true, a0, true, b, acc0[t], false, false);
            acc1[t] = __builtin_amdgcn_wmma_i32_16x16x64_iu8(
                true, a1, true, b, acc1[t], false, false);
        }
    }

    __syncthreads();

    #pragma unroll
    for (int t = 0; t < 8; ++t) {
        const int n  = n0 + t * 16 + lmod;
        const float sw = a_scale * wscale[n];
        const float bv = bias[n];
        #pragma unroll
        for (int r = 0; r < 8; ++r) {
            const int mA = r + (lhi << 3);
            const int mB = mA + 16;
            float zA = (float)acc0[t][r] * sw + bv
                     + res[(size_t)(m0 + mA) * H + n];
            float zB = (float)acc1[t][r] * sw + bv
                     + res[(size_t)(m0 + mB) * H + n];
            zsm[mA * LNS + n] = zA;
            zsm[mB * LNS + n] = zB;
        }
    }
    __syncthreads();

    #pragma unroll
    for (int rr = 0; rr < 4; ++rr) {
        const int row = wave * 4 + rr;
        float s = 0.0f, ss = 0.0f;
        for (int c = lane; c < H; c += 32) {
            float z = zsm[row * LNS + c];
            s += z; ss += z * z;
        }
        #pragma unroll
        for (int o = 16; o > 0; o >>= 1) {
            s  += __shfl_xor(s,  o, 32);
            ss += __shfl_xor(ss, o, 32);
        }
        const float mu  = s / (float)H;
        const float var = ss / (float)H - mu * mu;
        const float rs  = rsqrtf(var + LN_EPS);
        for (int c = lane; c < H; c += 32) {
            float z = zsm[row * LNS + c];
            out[(size_t)(m0 + row) * H + c] = (z - mu) * rs * gamma[c] + beta[c];
        }
    }
}

extern "C" void kernel_launch(void* const* d_in, const int* in_sizes, int n_in,
                              void* d_out, int out_size, void* d_ws, size_t ws_size,
                              hipStream_t stream) {
    const float* x     = (const float*)d_in[0];
    const float* res   = (const float*)d_in[1];
    const float* W     = (const float*)d_in[2];
    const float* bias  = (const float*)d_in[3];
    const float* gamma = (const float*)d_in[4];
    const float* beta  = (const float*)d_in[5];
    float* out = (float*)d_out;

    const int H = in_sizes[3];
    const long long M = (long long)in_sizes[0] / H;

    unsigned* absbits = (unsigned*)d_ws;
    float* wscale = (float*)((char*)d_ws + 256);
    size_t wq_off = 256 + (((size_t)H * 4 + 255) / 256) * 256;
    signed char* Wq = (signed char*)d_ws + wq_off;

    init_ws_kernel<<<1, 1, 0, stream>>>(absbits);
    absmax_kernel<<<2048, 256, 0, stream>>>(x, (size_t)M * H, absbits);
    quantw_kernel<<<H / 8, 256, 0, stream>>>(W, H, Wq, wscale);

    size_t shmem = (size_t)32 * (H + 4) * sizeof(float);
    if ((size_t)32 * H > shmem) shmem = (size_t)32 * H;
    gemm_ln_kernel<<<(int)(M / 32), 256, shmem, stream>>>(
        x, res, Wq, wscale, bias, gamma, beta, absbits, out, (int)M, H);
}
